// _SelfAttentionLayer_4286377361559
// MI455X (gfx1250) — matrix operand order repaired and checked by reading, not compiled, not run on hardware
//
#include <hip/hip_runtime.h>
#include <stdint.h>

typedef __attribute__((ext_vector_type(16))) _Float16 v16h;
typedef __attribute__((ext_vector_type(8)))  float    v8f;
typedef __attribute__((ext_vector_type(4)))  int      v4i;

#define B_ 8
#define S_ 2048
#define D_ 1024
#define H_ 64
#define SCALE_ 0.03125f   // D^-0.5 = 1/32 (reference scales by embedding dim)

union Frag {
  v16h     h;
  uint32_t u[8];
};

#if defined(__has_builtin)
#if __has_builtin(__builtin_amdgcn_global_load_async_to_lds_b128)
#define HAVE_ASYNC_LDS 1
#endif
#endif

typedef __attribute__((address_space(3))) v4i ldsv4i;

// 16-byte global -> LDS copy; async DMA path on gfx1250 toolchains that
// expose the builtin, synchronous fallback otherwise.
__device__ __forceinline__ void copy16_g2l(const void* g, void* l) {
#ifdef HAVE_ASYNC_LDS
  __builtin_amdgcn_global_load_async_to_lds_b128((v4i*)g, (ldsv4i*)l, 0, 0);
#else
  *(uint4*)l = *(const uint4*)g;
#endif
}

__device__ __forceinline__ void wait_async_lds() {
#ifdef HAVE_ASYNC_LDS
  asm volatile("s_wait_asynccnt 0x0" ::: "memory");
#endif
}

// ---------------------------------------------------------------------------
// Kernel 1: Wq/Wk/Wv f32 [1024][64] -> f16 transposed WT[3][64][1024]
// ---------------------------------------------------------------------------
__global__ void prep_weights(const float* __restrict__ Wq,
                             const float* __restrict__ Wk,
                             const float* __restrict__ Wv,
                             _Float16* __restrict__ WT) {
  int idx = blockIdx.x * blockDim.x + threadIdx.x;
  if (idx >= 3 * D_ * H_) return;
  int mat = idx / (D_ * H_);
  int rem = idx - mat * (D_ * H_);
  int k = rem >> 6;   // 0..1023
  int h = rem & 63;   // 0..63
  const float* W = (mat == 0) ? Wq : (mat == 1) ? Wk : Wv;
  WT[(size_t)mat * H_ * D_ + (size_t)h * D_ + k] = (_Float16)W[(size_t)k * H_ + h];
}

// ---------------------------------------------------------------------------
// Kernel 2: q = x@Wq, k = x@Wk, vT = (x@Wv)^T   (f16 outputs, f32 accumulate)
// ---------------------------------------------------------------------------
__global__ void qkv_proj(const float* __restrict__ x,
                         const _Float16* __restrict__ WT,
                         _Float16* __restrict__ qbuf,
                         _Float16* __restrict__ kbuf,
                         _Float16* __restrict__ vT) {
  const int lane = threadIdx.x & 31;
  const int wave = threadIdx.x >> 5;
  const int lm   = lane & 15;
  const int hf   = lane >> 4;
  const int tile = blockIdx.x * 4 + wave;   // 16-row tile id
  const int m0   = tile * 16;

  v8f acc[12] = {};
  const float* xrow = x + (size_t)(m0 + lm) * D_;

  for (int kb = 0; kb < D_; kb += 32) {
    // A fragment: 16x32 f16, lane=M, K contiguous per lane half (k = 16*hf + s)
    Frag a;
#pragma unroll
    for (int j = 0; j < 8; ++j) {
      int k0 = hf * 16 + 2 * j;
      float2 f = *(const float2*)(xrow + kb + k0);
      a.h[2 * j]     = (_Float16)f.x;
      a.h[2 * j + 1] = (_Float16)f.y;
    }
#pragma unroll
    for (int t = 0; t < 12; ++t) {
      int mat = t >> 2, nt = t & 3;
      const _Float16* wrow = WT + (size_t)mat * H_ * D_
                           + (size_t)(nt * 16 + lm) * D_ + kb + hf * 16;
      Frag bf;
#pragma unroll
      for (int j = 0; j < 8; ++j)
        bf.u[j] = *(const uint32_t*)(wrow + 2 * j);
      acc[t] = __builtin_amdgcn_wmma_f32_16x16x32_f16(
                   false, a.h, false, bf.h, (short)0, acc[t], false, false);
    }
  }

#pragma unroll
  for (int t = 0; t < 12; ++t) {
    int mat = t >> 2, nt = t & 3;
#pragma unroll
    for (int r = 0; r < 8; ++r) {
      int m = r + 8 * hf;
      int g = m0 + m;                 // flattened B*S row
      int hc = nt * 16 + lm;
      _Float16 val = (_Float16)acc[t][r];
      if (mat == 0)      qbuf[(size_t)g * H_ + hc] = val;
      else if (mat == 1) kbuf[(size_t)g * H_ + hc] = val;
      else {
        int b = g >> 11, s = g & (S_ - 1);
        vT[((size_t)b * H_ + hc) * S_ + s] = val;
      }
    }
  }
}

// ---------------------------------------------------------------------------
// Kernel 3: fused causal flash attention with async K/V staging in LDS.
// Block = 4 waves = 64 query rows; wave = 16 rows.  Key tiles of 64, kt<=qt.
// ---------------------------------------------------------------------------
__global__ void attn(const _Float16* __restrict__ qbuf,
                     const _Float16* __restrict__ kbuf,
                     const _Float16* __restrict__ vT,
                     float* __restrict__ out) {
  __shared__ __align__(16) _Float16 Kt[64][64];        // [key_rel][h]
  __shared__ __align__(16) _Float16 Vt[64][64];        // [h][key_rel]
  __shared__ __align__(16) _Float16 Pbuf[4][16][72];   // per-wave P (+pad)

  const int tid  = threadIdx.x;
  const int lane = tid & 31;
  const int wave = tid >> 5;
  const int lm   = lane & 15;
  const int hf   = lane >> 4;
  const int qt   = blockIdx.x;            // 0..31
  const int b    = blockIdx.y;            // 0..7
  const int qr0  = qt * 64 + wave * 16;

  // Q A-fragments (16x32 each, two h-steps) held in registers for all kt
  Frag qf[2];
#pragma unroll
  for (int st = 0; st < 2; ++st) {
    const _Float16* qrow = qbuf + (size_t)(b * S_ + qr0 + lm) * H_ + st * 32;
#pragma unroll
    for (int j = 0; j < 8; ++j) {
      int k0 = hf * 16 + 2 * j;
      qf[st].u[j] = *(const uint32_t*)(qrow + k0);
    }
  }

  float mrow[8], lsum[8];
#pragma unroll
  for (int r = 0; r < 8; ++r) { mrow[r] = -1e30f; lsum[r] = 0.0f; }
  v8f Oacc[4] = {};

  for (int kt = 0; kt <= qt; ++kt) {
    // ---- stage K tile (8KB contiguous) and V tile into LDS once per block
    __syncthreads();   // previous iteration's LDS reads complete
    {
      const _Float16* ksrc = kbuf + (size_t)(b * S_ + kt * 64) * H_;
      _Float16* kflat = &Kt[0][0];
#pragma unroll
      for (int i = 0; i < 4; ++i) {
        int c = tid + 128 * i;               // 16-byte chunk id, 512 total
        copy16_g2l(ksrc + c * 8, kflat + c * 8);
      }
#pragma unroll
      for (int i = 0; i < 4; ++i) {
        int h   = (tid >> 3) + 16 * i;       // 0..63
        int seg = (tid & 7) * 8;             // halfs within row
        copy16_g2l(vT + ((size_t)b * H_ + h) * S_ + kt * 64 + seg, &Vt[h][seg]);
      }
      // prefetch next key tile into cache while we compute on this one
      if (kt < qt)
        __builtin_prefetch(kbuf + (size_t)(b * S_ + (kt + 1) * 64) * H_ + tid * 32, 0, 0);
    }
    wait_async_lds();
    __syncthreads();   // tile visible to all waves

    // ---- S = Q @ K^T  (B-frag: lane = key row, K-pairs contiguous in Kt)
    v8f s4[4];
#pragma unroll
    for (int nt = 0; nt < 4; ++nt) {
      v8f sc = {};
#pragma unroll
      for (int st = 0; st < 2; ++st) {
        const _Float16* kp = &Kt[nt * 16 + lm][st * 32 + hf * 16];
        Frag kf;
#pragma unroll
        for (int j = 0; j < 8; ++j) kf.u[j] = *(const uint32_t*)(kp + 2 * j);
        sc = __builtin_amdgcn_wmma_f32_16x16x32_f16(
                 false, qf[st].h, false, kf.h, (short)0, sc, false, false);
      }
      s4[nt] = sc;
    }

    // ---- scale + causal mask (only diagonal tile needs masking)
    const bool diag = (kt == qt);
#pragma unroll
    for (int nt = 0; nt < 4; ++nt)
#pragma unroll
      for (int r = 0; r < 8; ++r) {
        float v = s4[nt][r] * SCALE_;
        if (diag) {
          int qg = qr0 + r + 8 * hf;
          int kg = kt * 64 + nt * 16 + lm;
          if (kg > qg) v = -1e30f;
        }
        s4[nt][r] = v;
      }

    // ---- online softmax: row max / rescale / row sum (butterfly in halves)
#pragma unroll
    for (int r = 0; r < 8; ++r) {
      float mx = fmaxf(fmaxf(s4[0][r], s4[1][r]), fmaxf(s4[2][r], s4[3][r]));
#pragma unroll
      for (int msk = 1; msk < 16; msk <<= 1)
        mx = fmaxf(mx, __shfl_xor(mx, msk, 32));
      float mn    = fmaxf(mrow[r], mx);
      float alpha = __expf(mrow[r] - mn);
      mrow[r] = mn;
      float rs = 0.0f;
#pragma unroll
      for (int nt = 0; nt < 4; ++nt) {
        float p = __expf(s4[nt][r] - mn);
        s4[nt][r] = p;
        rs += p;
      }
#pragma unroll
      for (int msk = 1; msk < 16; msk <<= 1)
        rs += __shfl_xor(rs, msk, 32);
      lsum[r] = lsum[r] * alpha + rs;
#pragma unroll
      for (int ht = 0; ht < 4; ++ht) Oacc[ht][r] *= alpha;
    }

    // ---- stage P (f16) in per-wave LDS region (same-wave LDS ops in order)
#pragma unroll
    for (int nt = 0; nt < 4; ++nt)
#pragma unroll
      for (int r = 0; r < 8; ++r)
        Pbuf[wave][r + 8 * hf][nt * 16 + lm] = (_Float16)s4[nt][r];
    asm volatile("s_wait_dscnt 0x0" ::: "memory");

    // ---- O += P @ V  (A-frag from LDS; B-frag contiguous in Vt)
#pragma unroll
    for (int st = 0; st < 2; ++st) {
      Frag pa;
#pragma unroll
      for (int j = 0; j < 8; ++j) {
        int kc = st * 32 + hf * 16 + 2 * j;
        pa.u[j] = *(const uint32_t*)&Pbuf[wave][lm][kc];
      }
#pragma unroll
      for (int ht = 0; ht < 4; ++ht) {
        Frag vf;
        const _Float16* vp = &Vt[ht * 16 + lm][st * 32 + hf * 16];
#pragma unroll
        for (int j = 0; j < 8; ++j) vf.u[j] = *(const uint32_t*)(vp + 2 * j);
        Oacc[ht] = __builtin_amdgcn_wmma_f32_16x16x32_f16(
                       false, pa.h, false, vf.h, (short)0, Oacc[ht], false, false);
      }
    }
  }

  // ---- normalize and write f32 output
#pragma unroll
  for (int ht = 0; ht < 4; ++ht)
#pragma unroll
    for (int r = 0; r < 8; ++r) {
      int m = r + 8 * hf;
      out[(size_t)(b * S_ + qr0 + m) * H_ + ht * 16 + lm] = Oacc[ht][r] / lsum[r];
    }
}

// ---------------------------------------------------------------------------
extern "C" void kernel_launch(void* const* d_in, const int* in_sizes, int n_in,
                              void* d_out, int out_size, void* d_ws, size_t ws_size,
                              hipStream_t stream) {
  (void)in_sizes; (void)n_in; (void)out_size; (void)ws_size;
  const float* x  = (const float*)d_in[0];
  const float* Wq = (const float*)d_in[1];
  const float* Wk = (const float*)d_in[2];
  const float* Wv = (const float*)d_in[3];
  float* out = (float*)d_out;

  _Float16* ws   = (_Float16*)d_ws;
  _Float16* qbuf = ws;                                   // [B*S][H] f16
  _Float16* kbuf = qbuf + (size_t)B_ * S_ * H_;          // [B*S][H] f16
  _Float16* vT   = kbuf + (size_t)B_ * S_ * H_;          // [B][H][S] f16
  _Float16* WT   = vT   + (size_t)B_ * S_ * H_;          // [3][H][D] f16

  prep_weights<<<(3 * D_ * H_ + 255) / 256, 256, 0, stream>>>(Wq, Wk, Wv, WT);
  qkv_proj<<<(B_ * S_ / 16) / 4, 128, 0, stream>>>(x, WT, qbuf, kbuf, vT);
  dim3 grid(S_ / 64, B_);
  attn<<<grid, 128, 0, stream>>>(qbuf, kbuf, vT, out);
}
